// BiLSTM_16707422781942
// MI455X (gfx1250) — matrix operand order repaired and checked by reading, not compiled, not run on hardware
//
#include <hip/hip_runtime.h>
#include <hip/hip_bf16.h>
#include <math.h>

// ---------------------------------------------------------------------------
// BiLSTM for MI455X (gfx1250).  T=512, B=64, E=H=1024.
// Per step/direction: gates[64,4096] = [x_t ; h][64,2048] @ W_cat^T + bias,
// via v_wmma_f32_16x16x32_bf16.  Weight tiles DMA'd into LDS with the Tensor
// Data Mover (tensor_load_to_lds, TENSORcnt), K split into 4 uniform-source
// phases so the 16 WMMA sub-iterations per phase run barrier-free.
// ---------------------------------------------------------------------------

#define T_LEN 512
#define B_SZ  64
#define E_SZ  1024
#define H_SZ  1024
#define K_TOT 2048          // E + H
#define NG    4096          // 4*H gate rows per direction
#define JB_N  32            // hidden-block count per direction (H/32)
#define ROWS_PER_WG 128     // 4 gates * 32 hidden units
#define PK    512           // K per phase
#define NPHASE 4

typedef __attribute__((ext_vector_type(16))) __bf16 bf16x16;
typedef __attribute__((ext_vector_type(8)))  float  f32x8;
typedef unsigned int uintx4 __attribute__((ext_vector_type(4)));
typedef int          intx4  __attribute__((ext_vector_type(4)));
typedef int          intx8  __attribute__((ext_vector_type(8)));

union FragU {
    uint4   u[2];
    bf16x16 v;
};

__device__ __forceinline__ unsigned short f2bf(float f) {
    unsigned int u = __float_as_uint(f);
    unsigned int r = u + 0x7FFFu + ((u >> 16) & 1u);   // round-to-nearest-even
    return (unsigned short)(r >> 16);
}

__device__ __forceinline__ float sigm(float x) {
    return 1.0f / (1.0f + __expf(-x));
}

// ---------------------------------------------------------------------------
// Setup: fp32 -> bf16 conversion of the input sequence. [T,B,E] row-major.
// ---------------------------------------------------------------------------
__global__ void setup_x_kernel(const float* __restrict__ inp,
                               unsigned short* __restrict__ x_bf) {
    const size_t nchunk = (size_t)T_LEN * B_SZ * E_SZ / 4;
    for (size_t i = (size_t)blockIdx.x * blockDim.x + threadIdx.x;
         i < nchunk; i += (size_t)gridDim.x * blockDim.x) {
        float4 f = *(const float4*)(inp + i * 4);
        unsigned short* o = x_bf + i * 4;
        o[0] = f2bf(f.x); o[1] = f2bf(f.y); o[2] = f2bf(f.z); o[3] = f2bf(f.w);
    }
}

// ---------------------------------------------------------------------------
// Setup: pack [W_ih | W_hh] (both directions) into bf16, rows grouped so each
// workgroup's 128 gate-rows are contiguous:
//   row R = ((d*32 + jb)*4 + q)*32 + c  <->  original gate row n = q*H+jb*32+c
//   cols k: [0,E) from W_ih, [E,K_TOT) from W_hh.
// ---------------------------------------------------------------------------
__global__ void setup_w_kernel(const float* __restrict__ W_ih_fw,
                               const float* __restrict__ W_hh_fw,
                               const float* __restrict__ W_ih_bw,
                               const float* __restrict__ W_hh_bw,
                               unsigned short* __restrict__ W_pack) {
    const size_t nchunk = (size_t)2 * NG * K_TOT / 4;
    for (size_t i = (size_t)blockIdx.x * blockDim.x + threadIdx.x;
         i < nchunk; i += (size_t)gridDim.x * blockDim.x) {
        size_t R  = i / (K_TOT / 4);
        int    kc = (int)(i % (K_TOT / 4)) * 4;
        int d   = (int)(R >> 12);
        int rem = (int)(R & 4095);
        int jb  = rem >> 7;
        int q   = (rem >> 5) & 3;
        int c   = rem & 31;
        int n   = q * H_SZ + jb * 32 + c;
        const float* src;
        if (kc < E_SZ) {
            const float* W = d ? W_ih_bw : W_ih_fw;
            src = W + (size_t)n * E_SZ + kc;
        } else {
            const float* W = d ? W_hh_bw : W_hh_fw;
            src = W + (size_t)n * H_SZ + (kc - E_SZ);
        }
        float4 f = *(const float4*)src;
        unsigned short* o = W_pack + R * K_TOT + kc;
        o[0] = f2bf(f.x); o[1] = f2bf(f.y); o[2] = f2bf(f.z); o[3] = f2bf(f.w);
    }
}

// ---------------------------------------------------------------------------
// Setup: packed bias (b_ih+b_hh, W_pack row order), initial h/c state, and
// the two init-valued rows of hid_out.
// ---------------------------------------------------------------------------
__global__ void setup_misc_kernel(const float* __restrict__ b_ih_fw,
                                  const float* __restrict__ b_hh_fw,
                                  const float* __restrict__ b_ih_bw,
                                  const float* __restrict__ b_hh_bw,
                                  const float* __restrict__ init_hidd,
                                  const float* __restrict__ init_cell,
                                  float* __restrict__ bias_pack,        // [2*NG]
                                  float* __restrict__ c_state,          // [2,B,H]
                                  unsigned short* __restrict__ h_bf,    // [2,B,H]
                                  float* __restrict__ hid_out) {        // [B,T+1,2H]
    const int NBIAS  = 2 * NG;
    const int NSTATE = 2 * B_SZ * H_SZ;
    const int NOUT   = 2 * B_SZ * H_SZ;
    const int total  = NBIAS + NSTATE + NOUT;
    for (int i = blockIdx.x * blockDim.x + threadIdx.x;
         i < total; i += gridDim.x * blockDim.x) {
        if (i < NBIAS) {
            int R = i, d = R >> 12, rem = R & 4095;
            int jb = rem >> 7, q = (rem >> 5) & 3, c = rem & 31;
            int n = q * H_SZ + jb * 32 + c;
            bias_pack[R] = d ? (b_ih_bw[n] + b_hh_bw[n])
                             : (b_ih_fw[n] + b_hh_fw[n]);
        } else if (i < NBIAS + NSTATE) {
            int idx = i - NBIAS;
            int r   = idx % (B_SZ * H_SZ);  // both directions share init state
            c_state[idx] = init_cell[r];
            h_bf[idx]    = f2bf(init_hidd[r]);
        } else {
            int idx = i - NBIAS - NSTATE;
            int which = idx / (B_SZ * H_SZ);
            int r  = idx % (B_SZ * H_SZ);
            int b  = r / H_SZ, j = r % H_SZ;
            float v = init_hidd[r];
            size_t bstride = (size_t)(T_LEN + 1) * 2 * H_SZ;
            if (which == 0)   // fw_states[T] = init_hidd
                hid_out[(size_t)b * bstride + (size_t)T_LEN * 2 * H_SZ + j] = v;
            else              // bw_states[0] = init_hidd
                hid_out[(size_t)b * bstride + H_SZ + j] = v;
        }
    }
}

// ---------------------------------------------------------------------------
// One recurrent step, both directions.  grid = 64 WGs (dir * 32 hidden-blocks)
// block = 256 threads = 8 waves.  Gates tile [64 batch x 128 cols].
// K loop: 4 phases of 512; per phase the B slab (128x512 bf16, 128KB) is
// DMA'd to LDS by the TDM while all threads stage the A slab (64x512, 64KB);
// then 16 barrier-free WMMA sub-iterations.
// ---------------------------------------------------------------------------
__global__ __launch_bounds__(256) void lstm_step_kernel(
        const unsigned short* __restrict__ x_bf,      // [T,B,E] bf16
        const unsigned short* __restrict__ W_pack,    // [2*NG, K_TOT] bf16
        const float*          __restrict__ bias_pack, // [2*NG]
        float*                __restrict__ c_state,   // [2,B,H]
        unsigned short*       __restrict__ h_bf,      // [2,B,H]
        float*                __restrict__ hid_out,   // [B,T+1,2H]
        float*                __restrict__ last_hidd, // [B,2H]
        float*                __restrict__ last_cell, // [B,2H]
        int step) {
    // 128KB B slab + 64KB A slab (aliased by 32KB gates after the GEMM)
    __shared__ __align__(16) unsigned char smem[(ROWS_PER_WG * PK + B_SZ * PK) * 2];
    unsigned short* B_slab = (unsigned short*)smem;                       // [128][512]
    unsigned short* A_slab = (unsigned short*)(smem + ROWS_PER_WG * PK * 2); // [64][512]
    float* gates_lds = (float*)(smem + ROWS_PER_WG * PK * 2);             // alias A

    const int d  = blockIdx.x >> 5;          // direction
    const int jb = blockIdx.x & 31;          // hidden block (32 units)
    const int rows_base = (d * JB_N + jb) * ROWS_PER_WG;
    const int xt = (d == 0) ? step : (T_LEN - 1 - step);

    const int tid  = threadIdx.x;
    const int lane = tid & 31;
    const int wid  = tid >> 5;
    const int wave_mrow = (wid & 3) * 16;
    const int wave_ncol = (wid >> 2) * 64;
    const int lhalf     = lane & 15;

    // Bias-initialized fp32 accumulators (all 8 elems of a lane share a col).
    f32x8 acc[4];
#pragma unroll
    for (int t4 = 0; t4 < 4; ++t4) {
        float b = bias_pack[rows_base + wave_ncol + t4 * 16 + lhalf];
        acc[t4] = (f32x8){b, b, b, b, b, b, b, b};
    }

    const unsigned lds_b = (unsigned)(size_t)B_slab;   // low 32 bits = LDS addr

    for (int p = 0; p < NPHASE; ++p) {
        __syncthreads();            // previous phase consumers done

#if __has_builtin(__builtin_amdgcn_tensor_load_to_lds)
        // ---- TDM: DMA the 128x512 bf16 weight tile into LDS (wave 0) ------
        if (wid == 0) {
            unsigned long long ga = (unsigned long long)(size_t)
                (W_pack + (size_t)rows_base * K_TOT + p * PK);
            uintx4 g0;
            g0.x = 1u;                                   // count=1, user D#
            g0.y = lds_b;                                // lds_addr
            g0.z = (unsigned)ga;                         // global_addr[31:0]
            g0.w = (unsigned)(ga >> 32) | (2u << 30);    // addr[56:32] | type=2
            intx8 g1;
            g1[0] = (int)(1u << 16);                     // data_size = 2 bytes
            g1[1] = (int)((unsigned)PK << 16);           // tensor_dim0 = 512
            g1[2] = (int)((unsigned)ROWS_PER_WG << 16);  // tensor_dim1 = 128
            g1[3] = (int)((unsigned)PK << 16);           // tile_dim0 = 512
            g1[4] = (int)ROWS_PER_WG;                    // tile_dim1 = 128
            g1[5] = (int)K_TOT;                          // dim0_stride = 2048
            g1[6] = 0;
            g1[7] = 0;
            intx4 g2 = (intx4){0, 0, 0, 0};
            intx4 g3 = (intx4){0, 0, 0, 0};
            intx8 g4 = (intx8){0, 0, 0, 0, 0, 0, 0, 0};  // unused extra group
            __builtin_amdgcn_tensor_load_to_lds(g0, g1, g2, g3, g4, 0);
        }
#else
        // ---- fallback: cooperative global->LDS staging of B slab ----------
        for (int it = 0; it < 32; ++it) {
            int chunk = tid + 256 * it;                  // 8192 uint4 chunks
            int nl   = chunk >> 6;
            int koff = (chunk & 63) * 8;
            *(uint4*)&B_slab[nl * PK + koff] =
                *(const uint4*)(W_pack + (size_t)(rows_base + nl) * K_TOT +
                                p * PK + koff);
        }
#endif

        // ---- stage A slab [64 rows x 512 k] (overlaps the DMA) ------------
        const unsigned short* abase =
            (p < 2) ? x_bf + (size_t)xt * B_SZ * E_SZ + p * PK
                    : h_bf + (size_t)d * B_SZ * H_SZ + (p - 2) * PK;
#pragma unroll
        for (int it = 0; it < 16; ++it) {
            int chunk = tid + 256 * it;                  // 4096 uint4 chunks
            int row  = chunk >> 6;
            int koff = (chunk & 63) * 8;
            *(uint4*)&A_slab[row * PK + koff] =
                *(const uint4*)(abase + (size_t)row * 1024 + koff);
        }

#if __has_builtin(__builtin_amdgcn_tensor_load_to_lds)
        if (wid == 0) __builtin_amdgcn_s_wait_tensorcnt(0);
#endif
        __syncthreads();            // A + B slabs visible to all waves

        // ---- 16 barrier-free WMMA sub-iterations --------------------------
        for (int kk = 0; kk < PK; kk += 32) {
            FragU a;
            {
                int row = wave_mrow + lhalf;
                int kb  = (lane < 16) ? 0 : 8;           // K 0..7 / 8..15
                a.u[0] = *(const uint4*)&A_slab[row * PK + kk + kb];
                a.u[1] = *(const uint4*)&A_slab[row * PK + kk + kb + 16];
            }
#pragma unroll
            for (int t4 = 0; t4 < 4; ++t4) {
                int nl = wave_ncol + t4 * 16 + lhalf;
                int kb = (lane < 16) ? 0 : 8;
                FragU bf;
                bf.u[0] = *(const uint4*)&B_slab[nl * PK + kk + kb];
                bf.u[1] = *(const uint4*)&B_slab[nl * PK + kk + kb + 16];
                acc[t4] = __builtin_amdgcn_wmma_f32_16x16x32_bf16(
                    false, a.v, false, bf.v, (short)0, acc[t4], false, false);
            }
        }
    }
    __syncthreads();                 // all A-slab reads done before aliasing

    // ---- spill accumulators to LDS per C/D 16x16 f32 layout --------------
#pragma unroll
    for (int t4 = 0; t4 < 4; ++t4) {
        int n     = wave_ncol + t4 * 16 + lhalf;
        int mbase = wave_mrow + ((lane < 16) ? 0 : 8);
#pragma unroll
        for (int r = 0; r < 8; ++r)
            gates_lds[(mbase + r) * ROWS_PER_WG + n] = acc[t4][r];
    }
    __syncthreads();

    // ---- elementwise LSTM cell update: 64 batch x 32 hidden --------------
    const size_t bstride = (size_t)(T_LEN + 1) * 2 * H_SZ;
#pragma unroll
    for (int u = 0; u < 8; ++u) {
        int idx = tid + 256 * u;             // 0..2047
        int b = idx >> 5;
        int c = idx & 31;
        float ig = gates_lds[b * ROWS_PER_WG +        c];
        float fg = gates_lds[b * ROWS_PER_WG +  32 +  c];
        float gg = gates_lds[b * ROWS_PER_WG +  64 +  c];
        float og = gates_lds[b * ROWS_PER_WG +  96 +  c];
        int j = jb * 32 + c;
        size_t sidx = ((size_t)d * B_SZ + b) * H_SZ + j;
        float cn = sigm(fg) * c_state[sidx] + sigm(ig) * tanhf(gg);
        float hn = sigm(og) * tanhf(cn);
        c_state[sidx] = cn;
        h_bf[sidx]    = f2bf(hn);
        int   row  = (d == 0) ? step : (T_LEN - step);
        int   cofs = (d == 0) ? 0 : H_SZ;
        hid_out[(size_t)b * bstride + (size_t)row * 2 * H_SZ + cofs + j] = hn;
        if (step == T_LEN - 1) {
            last_hidd[(size_t)b * 2 * H_SZ + cofs + j] = hn;
            last_cell[(size_t)b * 2 * H_SZ + cofs + j] = hn; // ref duplicates h
        }
    }
}

// ---------------------------------------------------------------------------
extern "C" void kernel_launch(void* const* d_in, const int* in_sizes, int n_in,
                              void* d_out, int out_size, void* d_ws, size_t ws_size,
                              hipStream_t stream) {
    const float* inp       = (const float*)d_in[0];
    const float* init_hidd = (const float*)d_in[1];
    const float* init_cell = (const float*)d_in[2];
    const float* W_ih_fw   = (const float*)d_in[3];
    const float* W_hh_fw   = (const float*)d_in[4];
    const float* b_ih_fw   = (const float*)d_in[5];
    const float* b_hh_fw   = (const float*)d_in[6];
    const float* W_ih_bw   = (const float*)d_in[7];
    const float* W_hh_bw   = (const float*)d_in[8];
    const float* b_ih_bw   = (const float*)d_in[9];
    const float* b_hh_bw   = (const float*)d_in[10];

    float* out       = (float*)d_out;
    float* hid_out   = out;
    float* last_hidd = out + (size_t)B_SZ * (T_LEN + 1) * 2 * H_SZ;
    float* last_cell = last_hidd + (size_t)B_SZ * 2 * H_SZ;

    // Workspace carve-out (~101.5 MB total)
    char* ws = (char*)d_ws;
    unsigned short* x_bf = (unsigned short*)ws;            // 64 MB
    ws += (size_t)T_LEN * B_SZ * E_SZ * sizeof(unsigned short);
    unsigned short* W_pack = (unsigned short*)ws;          // 32 MB
    ws += (size_t)2 * NG * K_TOT * sizeof(unsigned short);
    float* bias_pack = (float*)ws;                         // 32 KB
    ws += (size_t)2 * NG * sizeof(float);
    float* c_state = (float*)ws;                           // 512 KB
    ws += (size_t)2 * B_SZ * H_SZ * sizeof(float);
    unsigned short* h_bf = (unsigned short*)ws;            // 256 KB
    ws += (size_t)2 * B_SZ * H_SZ * sizeof(unsigned short);

    setup_x_kernel<<<2048, 256, 0, stream>>>(inp, x_bf);
    setup_w_kernel<<<2048, 256, 0, stream>>>(W_ih_fw, W_hh_fw, W_ih_bw, W_hh_bw,
                                             W_pack);
    setup_misc_kernel<<<1056, 256, 0, stream>>>(b_ih_fw, b_hh_fw, b_ih_bw, b_hh_bw,
                                                init_hidd, init_cell,
                                                bias_pack, c_state, h_bf, hid_out);

    for (int t = 0; t < T_LEN; ++t) {
        lstm_step_kernel<<<64, 256, 0, stream>>>(x_bf, W_pack, bias_pack,
                                                 c_state, h_bf,
                                                 hid_out, last_hidd, last_cell, t);
    }
}
